// MultiHeadAttention_48043504173472
// MI455X (gfx1250) — compile-verified, matrix operand order checked by reading, not run on hardware
//
#include <hip/hip_runtime.h>
#include <hip/hip_bf16.h>

// ---------------------------------------------------------------- constants
#define D_MODEL   1024
#define NUM_HEADS 16
#define HEAD_DIM  64      // D_MODEL / NUM_HEADS
#define SEQ       2048
#define BATCH     4
#define MTOT      (BATCH * SEQ)   // 8192 rows for the projection GEMMs

// ---------------------------------------------------------------- types
typedef __attribute__((ext_vector_type(16))) __bf16 v16bf;
typedef __attribute__((ext_vector_type(2)))  __bf16 bf16x2;
typedef __attribute__((ext_vector_type(8)))  float  v8f;
typedef __attribute__((ext_vector_type(4)))  int    v4i_t;

// ------------------------------------------------- async global->LDS copies
#if defined(__has_builtin)
#if __has_builtin(__builtin_amdgcn_global_load_async_to_lds_b128)
#define HAVE_ASYNC_B128 1
#else
#define HAVE_ASYNC_B128 0
#endif
#if __has_builtin(__builtin_amdgcn_global_load_async_to_lds_b32)
#define HAVE_ASYNC_B32 1
#else
#define HAVE_ASYNC_B32 0
#endif
#else
#define HAVE_ASYNC_B128 0
#define HAVE_ASYNC_B32 0
#endif

static __device__ __forceinline__ void cp_b128(void* lds, const void* g) {
#if HAVE_ASYNC_B128
    __builtin_amdgcn_global_load_async_to_lds_b128(
        (__attribute__((address_space(1))) v4i_t*)g,
        (__attribute__((address_space(3))) v4i_t*)lds, 0, 0);
#else
    *(float4*)lds = *(const float4*)g;
#endif
}

static __device__ __forceinline__ void cp_b32(void* lds, const void* g) {
#if HAVE_ASYNC_B32
    __builtin_amdgcn_global_load_async_to_lds_b32(
        (__attribute__((address_space(1))) int*)g,
        (__attribute__((address_space(3))) int*)lds, 0, 0);
#else
    *(unsigned int*)lds = *(const unsigned int*)g;
#endif
}

static __device__ __forceinline__ void async_wait() {
#if HAVE_ASYNC_B128 || HAVE_ASYNC_B32
#if __has_builtin(__builtin_amdgcn_s_wait_asynccnt)
    __builtin_amdgcn_s_wait_asynccnt(0);
#else
    asm volatile("s_wait_asynccnt 0x0" ::: "memory");
#endif
#endif
}

// ---------------------------------------------------------------- helpers
static __device__ __forceinline__ unsigned short f32_to_bf16(float x) {
    return __builtin_bit_cast(unsigned short, (__bf16)x);   // native cvt
}

// WMMA 16x32 bf16 fragment from a row-major bf16 row pointer.
// Lane layout (ISA 7.12.2): lane L: row = L&15, half = L>>4
//   VGPR v in 0..3 : K = half*8 + 2v (+0/+1)
//   VGPR v in 4..7 : K = 16 + half*8 + 2(v-4) (+0/+1)    (K-pairs contiguous)
static __device__ __forceinline__ v16bf build_frag(const unsigned short* rowptr, int half) {
    const unsigned int* p32 = (const unsigned int*)rowptr;
    v16bf f;
#pragma unroll
    for (int v = 0; v < 8; ++v) {
        int kb = (v < 4) ? (half * 8 + 2 * v) : (16 + half * 8 + 2 * (v - 4));
        bf16x2 pr = __builtin_bit_cast(bf16x2, p32[kb >> 1]);
        f[2 * v]     = pr[0];
        f[2 * v + 1] = pr[1];
    }
    return f;
}

// Same mapping, f32 source converted on the fly.
static __device__ __forceinline__ v16bf build_frag_f32(const float* rowptr, int half) {
    v16bf f;
#pragma unroll
    for (int v = 0; v < 8; ++v) {
        int kb = (v < 4) ? (half * 8 + 2 * v) : (16 + half * 8 + 2 * (v - 4));
        f[2 * v]     = (__bf16)rowptr[kb];
        f[2 * v + 1] = (__bf16)rowptr[kb + 1];
    }
    return f;
}

// Same mapping, bf16 elements spaced 2 apart (paired-transpose V layout).
static __device__ __forceinline__ v16bf build_frag_stride2(const unsigned short* p, int half) {
    v16bf f;
#pragma unroll
    for (int v = 0; v < 8; ++v) {
        int kb = (v < 4) ? (half * 8 + 2 * v) : (16 + half * 8 + 2 * (v - 4));
        f[2 * v]     = __builtin_bit_cast(__bf16, p[kb * 2]);
        f[2 * v + 1] = __builtin_bit_cast(__bf16, p[(kb + 1) * 2]);
    }
    return f;
}

static __device__ __forceinline__ v8f wmma_bf16(v16bf a, v16bf b, v8f c) {
    return __builtin_amdgcn_wmma_f32_16x16x32_bf16(
        false, a, false, b, (short)0, c, false, false);
}

// ================================================================ kernel 1
// Y[b,h,s,d] (bf16) = X[m,:] @ W[n,:]   (m=b*SEQ+s, n=h*64+d)
// 128x64 block tile, KC=32, 8 waves in 4x2 grid, 2x2 WMMA tiles per wave.
// Tiles staged in f32 via async b128 copies; converted in fragment build.
__global__ __launch_bounds__(256) void proj_gemm_kernel(
        const float* __restrict__ A, const float* __restrict__ W,
        unsigned short* __restrict__ Y) {
    __shared__ float Asf[128][36];   // row stride 144B (16B aligned)
    __shared__ float Wsf[64][36];

    const int m0   = blockIdx.x * 128;
    const int n0   = blockIdx.y * 64;
    const int tid  = threadIdx.x;
    const int lane = tid & 31, wave = tid >> 5;
    const int wm   = wave >> 1, wn = wave & 1;
    const int r    = lane & 15, half = lane >> 4;

    v8f acc[2][2] = {};

    for (int k0 = 0; k0 < D_MODEL; k0 += 32) {
        __syncthreads();
#pragma unroll
        for (int i = 0; i < 4; ++i) {                   // 128x32 f32 A tile
            int idx = tid + i * 256;                    // 1024 x b128
            int row = idx >> 3, c4 = (idx & 7) * 4;
            cp_b128(&Asf[row][c4], A + (size_t)(m0 + row) * D_MODEL + k0 + c4);
        }
#pragma unroll
        for (int i = 0; i < 2; ++i) {                   // 64x32 f32 W tile
            int idx = tid + i * 256;                    // 512 x b128
            int row = idx >> 3, c4 = (idx & 7) * 4;
            cp_b128(&Wsf[row][c4], W + (size_t)(n0 + row) * D_MODEL + k0 + c4);
        }
        if (k0 + 32 < D_MODEL) {                        // prefetch next chunk
            __builtin_prefetch(A + (size_t)(m0 + (tid >> 1)) * D_MODEL + k0 + 32, 0, 1);
            __builtin_prefetch(W + (size_t)(n0 + (tid & 63)) * D_MODEL + k0 + 32, 0, 1);
        }
        async_wait();
        __syncthreads();

        v16bf a0 = build_frag_f32(&Asf[wm * 32 + r][0],      half);
        v16bf a1 = build_frag_f32(&Asf[wm * 32 + 16 + r][0], half);
        v16bf b0 = build_frag_f32(&Wsf[wn * 32 + r][0],      half);
        v16bf b1 = build_frag_f32(&Wsf[wn * 32 + 16 + r][0], half);

        acc[0][0] = wmma_bf16(a0, b0, acc[0][0]);
        acc[0][1] = wmma_bf16(a0, b1, acc[0][1]);
        acc[1][0] = wmma_bf16(a1, b0, acc[1][0]);
        acc[1][1] = wmma_bf16(a1, b1, acc[1][1]);
    }

    // C/D layout: VGPR e -> M = 8*half + e, N = r
#pragma unroll
    for (int i = 0; i < 2; ++i)
#pragma unroll
        for (int j = 0; j < 2; ++j)
#pragma unroll
            for (int e = 0; e < 8; ++e) {
                int m = m0 + wm * 32 + i * 16 + 8 * half + e;
                int n = n0 + wn * 32 + j * 16 + r;
                int b = m >> 11, s = m & (SEQ - 1);
                int h = n >> 6,  d = n & (HEAD_DIM - 1);
                Y[(((size_t)b * NUM_HEADS + h) * SEQ + s) * HEAD_DIM + d] =
                    f32_to_bf16(acc[i][j][e]);
            }
}

// ================================================================ kernel 2
// One block = one (b,h) head, 16 query rows.
__global__ __launch_bounds__(256) void attention_kernel(
        const unsigned short* __restrict__ Q, const unsigned short* __restrict__ K,
        const unsigned short* __restrict__ V,
        float* __restrict__ attnW, float* __restrict__ ctx) {
    __shared__ float          sS[16][SEQ];        // 131072 B scores / exp()
    __shared__ unsigned short sQ[16][64];         //   2048 B
    __shared__ unsigned int   sVt32[32][132];     //  16896 B V pair-transposed
    __shared__ float          sCtx[2][16][64];    //   8192 B P*V partials
    __shared__ float          sInv[16];

    const int bh   = blockIdx.x >> 7;
    const int q0   = (blockIdx.x & 127) * 16;
    const int tid  = threadIdx.x;
    const int lane = tid & 31, wave = tid >> 5;
    const int r    = lane & 15, half = lane >> 4;

    // ---- Q tile: raw b128 copy (16 x 64 bf16 = 128 chunks) --------------
    const size_t baseQ = ((size_t)bh * SEQ + q0) * HEAD_DIM;
    if (tid < 128) {
        int row = tid >> 3, c = (tid & 7) * 8;
        cp_b128(&sQ[row][c], Q + baseQ + (size_t)row * HEAD_DIM + c);
    }
    async_wait();
    __syncthreads();

    v16bf aq0 = build_frag(&sQ[r][0],  half);
    v16bf aq1 = build_frag(&sQ[r][32], half);

    // ---- phase 1: scores (wave covers 256 kv cols) ----------------------
    const size_t baseK = (size_t)bh * SEQ * HEAD_DIM;
    for (int t = 0; t < 16; ++t) {
        int kv = wave * 256 + t * 16 + r;              // B-matrix: lane -> column
        const unsigned short* kp = K + baseK + (size_t)kv * HEAD_DIM;
        v16bf b0 = build_frag(kp,      half);
        v16bf b1 = build_frag(kp + 32, half);
        v8f acc = {};
        acc = wmma_bf16(aq0, b0, acc);
        acc = wmma_bf16(aq1, b1, acc);
#pragma unroll
        for (int e = 0; e < 8; ++e)
            sS[8 * half + e][wave * 256 + t * 16 + r] = acc[e] * 0.125f;
    }
    __syncthreads();

    // ---- phase 2: softmax (wave w owns rows 2w, 2w+1) -------------------
    float* wRowOut = attnW + ((size_t)bh * SEQ + q0) * SEQ;
#pragma unroll
    for (int rr = 0; rr < 2; ++rr) {
        int row = wave * 2 + rr;
        float mx = -3.402823466e38f;
        for (int c = lane; c < SEQ; c += 32) mx = fmaxf(mx, sS[row][c]);
#pragma unroll
        for (int off = 16; off > 0; off >>= 1) mx = fmaxf(mx, __shfl_xor(mx, off, 32));
        float sm = 0.f;
        for (int c = lane; c < SEQ; c += 32) {
            float e = __expf(sS[row][c] - mx);
            sS[row][c] = e;                            // keep exp for P*V
            sm += e;
        }
#pragma unroll
        for (int off = 16; off > 0; off >>= 1) sm += __shfl_xor(sm, off, 32);
        float inv = 1.0f / sm;
        if (lane == 0) sInv[row] = inv;
        for (int c = lane; c < SEQ; c += 32)           // 1 GB weight stream
            wRowOut[(size_t)row * SEQ + c] = sS[row][c] * inv;
    }
    __syncthreads();

    // ---- phase 3: P*V  (wave -> ntile of HEAD_DIM, khalf of 128-kv chunk)
    const int ntile = wave & 3, khalf = wave >> 2;
    const int nidx  = ntile * 16 + r;
    v8f accv = {};
    const size_t baseV = (size_t)bh * SEQ * HEAD_DIM;
    for (int s = 0; s < 16; ++s) {                     // 16 chunks of 128 kv
        __syncthreads();
        // stage V chunk: dword (= dim-pair) scatter into pair-transposed LDS
#pragma unroll
        for (int i = 0; i < 16; ++i) {
            int idx = tid + i * 256;                   // 4096 dwords
            int kv  = idx >> 5;                        // 0..127
            int d2  = idx & 31;                        // dim-pair index
            cp_b32(&sVt32[d2][kv],
                   (const unsigned int*)(V + baseV + (size_t)(s * 128 + kv) * HEAD_DIM) + d2);
        }
        async_wait();
        __syncthreads();
#pragma unroll
        for (int kc = 0; kc < 2; ++kc) {
            int kloc = khalf * 64 + kc * 32;
            v16bf a = build_frag_f32(&sS[r][s * 128 + kloc], half);
            const unsigned short* vp =
                (const unsigned short*)&sVt32[nidx >> 1][kloc] + (nidx & 1);
            v16bf b = build_frag_stride2(vp, half);
            accv = wmma_bf16(a, b, accv);
        }
    }
#pragma unroll
    for (int e = 0; e < 8; ++e)
        sCtx[khalf][8 * half + e][nidx] = accv[e];
    __syncthreads();

    // reduce K-slices, apply 1/rowsum, write ctx [B,H,S,64] f32
#pragma unroll
    for (int i = 0; i < 4; ++i) {
        int idx = tid + i * 256;
        int m = idx >> 6, n = idx & 63;
        float val = (sCtx[0][m][n] + sCtx[1][m][n]) * sInv[m];
        ctx[((size_t)bh * SEQ + q0 + m) * HEAD_DIM + n] = val;
    }
}

// ================================================================ kernel 3
// Out[m,n] = concat(ctx)[m,:] @ Wo[n,:] + bo[n]   (f32 out)
__global__ __launch_bounds__(256) void out_gemm_kernel(
        const float* __restrict__ Ctx, const float* __restrict__ W,
        const float* __restrict__ bias, float* __restrict__ Out) {
    __shared__ float Asf[128][36];
    __shared__ float Wsf[64][36];

    const int m0   = blockIdx.x * 128;
    const int n0   = blockIdx.y * 64;
    const int tid  = threadIdx.x;
    const int lane = tid & 31, wave = tid >> 5;
    const int wm   = wave >> 1, wn = wave & 1;
    const int r    = lane & 15, half = lane >> 4;

    v8f acc[2][2] = {};

    for (int k0 = 0; k0 < D_MODEL; k0 += 32) {
        // concat gather: a 32-wide K chunk stays inside one head's 64 dims
        const int h  = k0 >> 6;
        const int d0 = k0 & (HEAD_DIM - 1);
        __syncthreads();
#pragma unroll
        for (int i = 0; i < 4; ++i) {
            int idx = tid + i * 256;
            int row = idx >> 3, c4 = (idx & 7) * 4;
            int m = m0 + row;
            int b = m >> 11, s = m & (SEQ - 1);
            cp_b128(&Asf[row][c4],
                    Ctx + (((size_t)b * NUM_HEADS + h) * SEQ + s) * HEAD_DIM + d0 + c4);
        }
#pragma unroll
        for (int i = 0; i < 2; ++i) {
            int idx = tid + i * 256;
            int row = idx >> 3, c4 = (idx & 7) * 4;
            cp_b128(&Wsf[row][c4], W + (size_t)(n0 + row) * D_MODEL + k0 + c4);
        }
        async_wait();
        __syncthreads();

        v16bf a0 = build_frag_f32(&Asf[wm * 32 + r][0],      half);
        v16bf a1 = build_frag_f32(&Asf[wm * 32 + 16 + r][0], half);
        v16bf b0 = build_frag_f32(&Wsf[wn * 32 + r][0],      half);
        v16bf b1 = build_frag_f32(&Wsf[wn * 32 + 16 + r][0], half);

        acc[0][0] = wmma_bf16(a0, b0, acc[0][0]);
        acc[0][1] = wmma_bf16(a0, b1, acc[0][1]);
        acc[1][0] = wmma_bf16(a1, b0, acc[1][0]);
        acc[1][1] = wmma_bf16(a1, b1, acc[1][1]);
    }

#pragma unroll
    for (int i = 0; i < 2; ++i)
#pragma unroll
        for (int j = 0; j < 2; ++j)
#pragma unroll
            for (int e = 0; e < 8; ++e) {
                int m = m0 + wm * 32 + i * 16 + 8 * half + e;
                int n = n0 + wn * 32 + j * 16 + r;
                Out[(size_t)m * D_MODEL + n] = acc[i][j][e] + bias[n];
            }
}

// ================================================================ launch
extern "C" void kernel_launch(void* const* d_in, const int* in_sizes, int n_in,
                              void* d_out, int out_size, void* d_ws, size_t ws_size,
                              hipStream_t stream) {
    const float* q  = (const float*)d_in[0];
    const float* k  = (const float*)d_in[1];
    const float* v  = (const float*)d_in[2];
    const float* Wq = (const float*)d_in[3];
    const float* Wk = (const float*)d_in[4];
    const float* Wv = (const float*)d_in[5];
    const float* Wo = (const float*)d_in[6];
    const float* bo = (const float*)d_in[7];

    float* out   = (float*)d_out;                                   // [B,S,D] f32
    float* attnw = out + (size_t)BATCH * SEQ * D_MODEL;             // [B,H,S,S] f32

    const size_t ELEMS = (size_t)BATCH * SEQ * D_MODEL;             // 8.39M
    char* ws = (char*)d_ws;
    unsigned short* qb  = (unsigned short*)ws;                      // bf16 [B,H,S,64]
    unsigned short* kb  = (unsigned short*)(ws + ELEMS * 2);
    unsigned short* vb  = (unsigned short*)(ws + ELEMS * 4);
    float*          ctx = (float*)(ws + ELEMS * 6);                 // f32 [B,H,S,64]

    dim3 gGemm(MTOT / 128, D_MODEL / 64);   // 64 x 16
    dim3 blk(256);

    proj_gemm_kernel<<<gGemm, blk, 0, stream>>>(q, Wq, qb);
    proj_gemm_kernel<<<gGemm, blk, 0, stream>>>(k, Wk, kb);
    proj_gemm_kernel<<<gGemm, blk, 0, stream>>>(v, Wv, vb);

    attention_kernel<<<dim3(BATCH * NUM_HEADS * (SEQ / 16)), blk, 0, stream>>>(
        qb, kb, vb, attnw, ctx);

    out_gemm_kernel<<<gGemm, blk, 0, stream>>>(ctx, Wo, bo, out);
}
